// SelfAttention_3891240370898
// MI455X (gfx1250) — matrix operand order repaired and checked by reading, not compiled, not run on hardware
//
#include <hip/hip_runtime.h>
#include <hip/hip_bf16.h>

#define BB     8
#define LL     2048
#define DMODEL 128
#define NH     4
#define DKH    32
#define NEGV   (-1e30f)

typedef __attribute__((ext_vector_type(16))) __bf16 v16bf;
typedef __attribute__((ext_vector_type(8)))  __bf16 v8bf;
typedef __attribute__((ext_vector_type(8)))  float  v8f;

__device__ __forceinline__ v8f wmma_bf16(v16bf a, v16bf b, v8f c) {
  return __builtin_amdgcn_wmma_f32_16x16x32_bf16(false, a, false, b, (short)0, c,
                                                 false, false);
}
__device__ __forceinline__ v16bf catbf(v8bf a, v8bf b) {
  return __builtin_shufflevector(a, b, 0,1,2,3,4,5,6,7,8,9,10,11,12,13,14,15);
}
__device__ __forceinline__ __bf16 f2bf(float f) { return (__bf16)f; }
__device__ __forceinline__ v8bf pack8(v8f v) {
  v8bf r;
  #pragma unroll
  for (int i = 0; i < 8; ++i) r[i] = (__bf16)v[i];
  return r;
}
// CDNA5 async global->LDS copy of 16 bytes (tracked by ASYNCcnt).
// lds_off = workgroup-relative LDS byte offset (low 32 bits of generic ptr).
__device__ __forceinline__ void async_copy_b128(unsigned lds_off, const void* gaddr) {
  asm volatile("global_load_async_to_lds_b128 %0, %1, off"
               :: "v"(lds_off), "v"(gaddr) : "memory");
}
__device__ __forceinline__ unsigned lds_off_of(const void* p) {
  return (unsigned)(unsigned long long)p;
}

// ---------------------------------------------------------------------------
// Kernel 1: convert Wq/Wk/Wv (H,D,dk) -> bf16 transposed (H,dk,D); Wo (D,D) ->
// bf16 transposed Wot[c][d] = Wo[d][c]. Transposed so WMMA B-operand loads are
// 16 contiguous bf16 per lane.
// ---------------------------------------------------------------------------
__global__ void convert_weights(const float* __restrict__ Wq, const float* __restrict__ Wk,
                                const float* __restrict__ Wv, const float* __restrict__ Wo,
                                __bf16* __restrict__ Wqt, __bf16* __restrict__ Wkt,
                                __bf16* __restrict__ Wvt, __bf16* __restrict__ Wot) {
  int tid = blockIdx.x * blockDim.x + threadIdx.x;
  const int wsz = NH * DMODEL * DKH;          // 16384
  if (tid < 3 * wsz) {
    int which = tid / wsz;
    int r = tid % wsz;
    int h = r / (DMODEL * DKH);
    int r2 = r % (DMODEL * DKH);
    int d = r2 / DKH;
    int c = r2 % DKH;
    const float* W = (which == 0) ? Wq : (which == 1) ? Wk : Wv;
    __bf16* Wt = (which == 0) ? Wqt : (which == 1) ? Wkt : Wvt;
    Wt[(h * DKH + c) * DMODEL + d] = f2bf(W[(h * DMODEL + d) * DKH + c]);
  } else {
    int r = tid - 3 * wsz;                    // over DMODEL*DMODEL
    int d = r / DMODEL, c = r % DMODEL;
    Wot[c * DMODEL + d] = f2bf(Wo[d * DMODEL + c]);
  }
}

// ---------------------------------------------------------------------------
// Kernel 2: x (B, D, L) fp32 -> xt (B, L, D) bf16 via LDS-tiled transpose.
// ---------------------------------------------------------------------------
__global__ void transpose_convert(const float* __restrict__ x, __bf16* __restrict__ xt) {
  __shared__ float tile[32][33];
  const int lb = blockIdx.x * 32;     // l tile
  const int db = blockIdx.y * 32;     // d tile
  const int b  = blockIdx.z;
  const int tx = threadIdx.x, ty = threadIdx.y;   // 32 x 8
  const float* src = x + (size_t)b * DMODEL * LL;
  #pragma unroll
  for (int i = 0; i < 32; i += 8)
    tile[ty + i][tx] = src[(size_t)(db + ty + i) * LL + lb + tx];
  __syncthreads();
  __bf16* dst = xt + (size_t)b * LL * DMODEL;
  #pragma unroll
  for (int i = 0; i < 32; i += 8)
    dst[(size_t)(lb + ty + i) * DMODEL + db + tx] = f2bf(tile[tx][ty + i]);
}

// ---------------------------------------------------------------------------
// Kernel 3: QKV projection. One wave per 16-row l-tile per (b,h).
// Q,K stored bf16 [B][H][L][32]; V stored transposed bf16 [B][H][32][L].
// ---------------------------------------------------------------------------
__global__ __launch_bounds__(128) void qkv_kernel(
    const __bf16* __restrict__ xt,
    const __bf16* __restrict__ Wqt, const __bf16* __restrict__ Wkt,
    const __bf16* __restrict__ Wvt,
    __bf16* __restrict__ Qb, __bf16* __restrict__ Kb, __bf16* __restrict__ Vt) {
  const int lane = threadIdx.x & 31, wvid = threadIdx.x >> 5;
  const int lo = lane & 15, hi = lane >> 4;
  int gid = blockIdx.x;
  const int lblk = gid % (LL / 64); gid /= (LL / 64);
  const int h = gid % NH; const int b = gid / NH;
  const int l0 = lblk * 64 + wvid * 16;

  const __bf16* arow = xt + ((size_t)b * LL + l0 + lo) * DMODEL;
  const __bf16* wq = Wqt + (size_t)h * DKH * DMODEL;
  const __bf16* wk = Wkt + (size_t)h * DKH * DMODEL;
  const __bf16* wvp = Wvt + (size_t)h * DKH * DMODEL;

  v8f q0a{}, q1a{}, k0a{}, k1a{}, v0a{}, v1a{};
  #pragma unroll
  for (int d0 = 0; d0 < DMODEL; d0 += 32) {
    v16bf a = catbf(*(const v8bf*)(arow + d0 + 16 * hi),
                    *(const v8bf*)(arow + d0 + 16 * hi + 8));
    v16bf bq0 = *(const v16bf*)(wq  + (size_t)lo        * DMODEL + d0 + 16 * hi);
    v16bf bq1 = *(const v16bf*)(wq  + (size_t)(16 + lo) * DMODEL + d0 + 16 * hi);
    v16bf bk0 = *(const v16bf*)(wk  + (size_t)lo        * DMODEL + d0 + 16 * hi);
    v16bf bk1 = *(const v16bf*)(wk  + (size_t)(16 + lo) * DMODEL + d0 + 16 * hi);
    v16bf bv0 = *(const v16bf*)(wvp + (size_t)lo        * DMODEL + d0 + 16 * hi);
    v16bf bv1 = *(const v16bf*)(wvp + (size_t)(16 + lo) * DMODEL + d0 + 16 * hi);
    q0a = wmma_bf16(a, bq0, q0a);
    q1a = wmma_bf16(a, bq1, q1a);
    k0a = wmma_bf16(a, bk0, k0a);
    k1a = wmma_bf16(a, bk1, k1a);
    v0a = wmma_bf16(a, bv0, v0a);
    v1a = wmma_bf16(a, bv1, v1a);
  }
  const size_t base = ((size_t)(b * NH + h) * LL + l0) * DKH;
  #pragma unroll
  for (int r = 0; r < 8; ++r) {
    const int row = r + 8 * hi;
    Qb[base + (size_t)row * DKH + lo]      = f2bf(q0a[r]);
    Qb[base + (size_t)row * DKH + 16 + lo] = f2bf(q1a[r]);
    Kb[base + (size_t)row * DKH + lo]      = f2bf(k0a[r]);
    Kb[base + (size_t)row * DKH + 16 + lo] = f2bf(k1a[r]);
  }
  __bf16* vt = Vt + (size_t)(b * NH + h) * DKH * LL;
  *(v8bf*)(vt + (size_t)lo        * LL + l0 + 8 * hi) = pack8(v0a);
  *(v8bf*)(vt + (size_t)(16 + lo) * LL + l0 + 8 * hi) = pack8(v1a);
}

// ---------------------------------------------------------------------------
// Kernel 4: flash attention. One wave per 16-query tile per (b,h); the 4 waves
// of a block share (b,h) and consume the same K/V stream, so K/V tiles are
// staged once per block into LDS with CDNA5 async global->LDS copies,
// double-buffered (prefetch kt+1 while computing kt, ASYNCcnt-throttled).
// head bf16 [B][L][128] output (heads concatenated).
// ---------------------------------------------------------------------------
__global__ __launch_bounds__(128) void attn_kernel(
    const __bf16* __restrict__ Qb, const __bf16* __restrict__ Kb,
    const __bf16* __restrict__ Vt, const float* __restrict__ mask,
    __bf16* __restrict__ head) {
  __shared__ __bf16 kbuf[2][32 * 32];   // [key 0..31][dk 0..31]
  __shared__ __bf16 vbuf[2][32 * 32];   // [dk 0..31][key 0..31]
  __shared__ __bf16 pbuf[4][16 * 32];
  const int tid = threadIdx.x;
  const int lane = tid & 31, wvid = tid >> 5;
  const int lo = lane & 15, hi = lane >> 4;
  int gid = blockIdx.x;
  const int lblk = gid % (LL / 64); gid /= (LL / 64);
  const int h = gid % NH; const int b = gid / NH;
  const int q0 = lblk * 64 + wvid * 16;

  const size_t bh = (size_t)(b * NH + h);
  const __bf16* qrow = Qb + (bh * LL + q0 + lo) * DKH;
  const v16bf qa = catbf(*(const v8bf*)(qrow + 16 * hi),
                         *(const v8bf*)(qrow + 16 * hi + 8));

  float m[8], l[8];
  v8f o0{}, o1{};
  #pragma unroll
  for (int r = 0; r < 8; ++r) { m[r] = -__builtin_inff(); l[r] = 0.f; }

  const float scale = 0.17677669529663687f;  // 1/sqrt(32)
  const char* kbytes = (const char*)(Kb + bh * LL * DKH);
  const char* vbytes = (const char*)(Vt + bh * (size_t)DKH * LL);
  const float* mrow = mask + (size_t)b * LL;
  __bf16* pw = &pbuf[wvid][0];

  // Per-thread async-copy slices of one 32x32 bf16 K tile and V tile.
  const int vrowi = tid >> 2;            // 0..31 (dk row of V tile)
  const int vcol16 = (tid & 3) * 16;     // 16B chunk within 64B row
  const unsigned kdst0 = lds_off_of(&kbuf[0][0]) + (unsigned)tid * 16u;
  const unsigned kdst1 = lds_off_of(&kbuf[1][0]) + (unsigned)tid * 16u;
  const unsigned vdst0 = lds_off_of(&vbuf[0][0]) + (unsigned)(vrowi * 64 + vcol16);
  const unsigned vdst1 = lds_off_of(&vbuf[1][0]) + (unsigned)(vrowi * 64 + vcol16);

  // Preload tile 0 into buffer 0.
  async_copy_b128(kdst0, kbytes + (size_t)tid * 16);
  async_copy_b128(vdst0, vbytes + (size_t)vrowi * (LL * 2) + vcol16);

  for (int kt = 0; kt < LL / 32; ++kt) {
    const int cur = kt & 1;
    const int k0 = kt * 32;
    // All waves done reading the buffer we are about to overwrite.
    __syncthreads();
    // Prefetch tile kt+1 (index wraps; tail prefetch is harmless, keeps the
    // ASYNCcnt wait value constant).
    {
      const int kn = ((kt + 1) & (LL / 32 - 1)) * 32;
      async_copy_b128(cur ? kdst0 : kdst1, kbytes + (size_t)kn * 64 + (size_t)tid * 16);
      async_copy_b128(cur ? vdst0 : vdst1,
                      vbytes + ((size_t)vrowi * LL + kn) * 2 + vcol16);
    }
    // Wait for current tile (leave the 2 prefetch copies in flight), publish.
    asm volatile("s_wait_asynccnt 0x2" ::: "memory");
    __syncthreads();

    // ---- S = Q * K^T (two 16-key subtiles) from LDS ----
    const __bf16* kb_ = &kbuf[cur][0];
    v16bf kb0 = *(const v16bf*)(kb_ + lo * 32 + 16 * hi);
    v16bf kb1 = *(const v16bf*)(kb_ + (16 + lo) * 32 + 16 * hi);
    v8f z{};
    v8f s0 = wmma_bf16(qa, kb0, z);
    v8f s1 = wmma_bf16(qa, kb1, z);
    // ---- scale + key mask (mask==1 => NEG) ----
    const float mk0 = mrow[k0 + lo];
    const float mk1 = mrow[k0 + 16 + lo];
    #pragma unroll
    for (int r = 0; r < 8; ++r) {
      s0[r] = (mk0 > 0.5f) ? NEGV : s0[r] * scale;
      s1[r] = (mk1 > 0.5f) ? NEGV : s1[r] * scale;
    }
    // ---- online softmax: row max over 32 keys (half-wave shfl reduce) ----
    float rmx[8];
    #pragma unroll
    for (int r = 0; r < 8; ++r) rmx[r] = fmaxf(s0[r], s1[r]);
    #pragma unroll
    for (int d = 1; d < 16; d <<= 1) {
      #pragma unroll
      for (int r = 0; r < 8; ++r) rmx[r] = fmaxf(rmx[r], __shfl_xor(rmx[r], d, 32));
    }
    float p0[8], p1[8], rs[8];
    #pragma unroll
    for (int r = 0; r < 8; ++r) {
      const float mn = fmaxf(m[r], rmx[r]);
      const float al = __expf(m[r] - mn);
      m[r] = mn;
      p0[r] = __expf(s0[r] - mn);
      p1[r] = __expf(s1[r] - mn);
      rs[r] = p0[r] + p1[r];
      l[r] *= al;
      o0[r] *= al;
      o1[r] *= al;
    }
    #pragma unroll
    for (int d = 1; d < 16; d <<= 1) {
      #pragma unroll
      for (int r = 0; r < 8; ++r) rs[r] += __shfl_xor(rs[r], d, 32);
    }
    #pragma unroll
    for (int r = 0; r < 8; ++r) l[r] += rs[r];
    // ---- D-layout -> A-layout via LDS (wave-private region) ----
    #pragma unroll
    for (int r = 0; r < 8; ++r) {
      const int row = r + 8 * hi;
      pw[row * 32 + lo]      = f2bf(p0[r]);
      pw[row * 32 + 16 + lo] = f2bf(p1[r]);
    }
    const __bf16* pr = pw + lo * 32;
    v16bf pa = catbf(*(const v8bf*)(pr + 16 * hi),
                     *(const v8bf*)(pr + 16 * hi + 8));
    // ---- O += P * V from LDS ----
    const __bf16* vb_ = &vbuf[cur][0];
    v16bf vb0 = *(const v16bf*)(vb_ + lo * 32 + 16 * hi);
    v16bf vb1 = *(const v16bf*)(vb_ + (16 + lo) * 32 + 16 * hi);
    o0 = wmma_bf16(pa, vb0, o0);
    o1 = wmma_bf16(pa, vb1, o1);
  }
  // ---- epilogue: normalize, apply query mask, store bf16 heads ----
  #pragma unroll
  for (int r = 0; r < 8; ++r) {
    const int row = r + 8 * hi;
    const int q = q0 + row;
    const float inv = mrow[q] / l[r];
    __bf16* hrow = head + ((size_t)b * LL + q) * DMODEL + h * DKH;
    hrow[lo]      = f2bf(o0[r] * inv);
    hrow[16 + lo] = f2bf(o1[r] * inv);
  }
}

// ---------------------------------------------------------------------------
// Kernel 5: output projection head (B,L,128) @ Wo -> out (B,128,L) fp32.
// ---------------------------------------------------------------------------
__global__ __launch_bounds__(128) void oproj_kernel(
    const __bf16* __restrict__ head, const __bf16* __restrict__ Wot,
    float* __restrict__ out) {
  const int lane = threadIdx.x & 31, wvid = threadIdx.x >> 5;
  const int lo = lane & 15, hi = lane >> 4;
  int gid = blockIdx.x;
  const int lblk = gid % (LL / 64);
  const int b = gid / (LL / 64);
  const int l0 = lblk * 64 + wvid * 16;
  const __bf16* arow = head + ((size_t)b * LL + l0 + lo) * DMODEL;
  v8f acc[8];
  #pragma unroll
  for (int t = 0; t < 8; ++t) acc[t] = v8f{};
  #pragma unroll
  for (int c0 = 0; c0 < DMODEL; c0 += 32) {
    v16bf a = catbf(*(const v8bf*)(arow + c0 + 16 * hi),
                    *(const v8bf*)(arow + c0 + 16 * hi + 8));
    #pragma unroll
    for (int t = 0; t < 8; ++t) {
      v16bf bt = *(const v16bf*)(Wot + (size_t)(16 * t + lo) * DMODEL + c0 + 16 * hi);
      acc[t] = wmma_bf16(a, bt, acc[t]);
    }
  }
  #pragma unroll
  for (int t = 0; t < 8; ++t) {
    float* ob = out + ((size_t)b * DMODEL + 16 * t + lo) * LL + l0 + 8 * hi;
    *(float4*)ob       = make_float4(acc[t][0], acc[t][1], acc[t][2], acc[t][3]);
    *(float4*)(ob + 4) = make_float4(acc[t][4], acc[t][5], acc[t][6], acc[t][7]);
  }
}

// ---------------------------------------------------------------------------
extern "C" void kernel_launch(void* const* d_in, const int* in_sizes, int n_in,
                              void* d_out, int out_size, void* d_ws, size_t ws_size,
                              hipStream_t stream) {
  (void)in_sizes; (void)n_in; (void)out_size; (void)ws_size;
  const float* x    = (const float*)d_in[0];
  const float* mask = (const float*)d_in[1];
  const float* Wq   = (const float*)d_in[2];
  const float* Wk   = (const float*)d_in[3];
  const float* Wv   = (const float*)d_in[4];
  const float* Wo   = (const float*)d_in[5];
  float* out = (float*)d_out;

  char* ws = (char*)d_ws;
  size_t off = 0;
  auto alloc = [&](size_t bytes) -> void* {
    void* p = ws + off;
    off += (bytes + 255) & ~(size_t)255;
    return p;
  };
  __bf16* xt   = (__bf16*)alloc((size_t)BB * LL * DMODEL * 2);
  __bf16* Qb   = (__bf16*)alloc((size_t)BB * NH * LL * DKH * 2);
  __bf16* Kb   = (__bf16*)alloc((size_t)BB * NH * LL * DKH * 2);
  __bf16* Vt   = (__bf16*)alloc((size_t)BB * NH * DKH * LL * 2);
  __bf16* head = (__bf16*)alloc((size_t)BB * LL * DMODEL * 2);
  __bf16* Wqt  = (__bf16*)alloc((size_t)NH * DKH * DMODEL * 2);
  __bf16* Wkt  = (__bf16*)alloc((size_t)NH * DKH * DMODEL * 2);
  __bf16* Wvt  = (__bf16*)alloc((size_t)NH * DKH * DMODEL * 2);
  __bf16* Wot  = (__bf16*)alloc((size_t)DMODEL * DMODEL * 2);

  convert_weights<<<256, 256, 0, stream>>>(Wq, Wk, Wv, Wo, Wqt, Wkt, Wvt, Wot);
  transpose_convert<<<dim3(LL / 32, DMODEL / 32, BB), dim3(32, 8), 0, stream>>>(x, xt);
  qkv_kernel<<<BB * NH * (LL / 64), 128, 0, stream>>>(xt, Wqt, Wkt, Wvt, Qb, Kb, Vt);
  attn_kernel<<<BB * NH * (LL / 64), 128, 0, stream>>>(Qb, Kb, Vt, mask, head);
  oproj_kernel<<<BB * (LL / 64), 128, 0, stream>>>(head, Wot, out);
}
